// DemandTemporalEncoder_36447092474236
// MI455X (gfx1250) — matrix operand order repaired and checked by reading, not compiled, not run on hardware
//
#include <hip/hip_runtime.h>
#include <hip/hip_bf16.h>
#include <math.h>

// ---------------- types for WMMA ----------------
typedef __attribute__((ext_vector_type(16))) __bf16 v16bf;
typedef __attribute__((ext_vector_type(8)))  float  v8f;
typedef unsigned short u16;
typedef __attribute__((ext_vector_type(8))) u16 u16x8;
typedef __attribute__((address_space(3))) u16 lds_u16;

union ABf {
    v16bf v;
    u16x8 h[2];
};

__device__ __forceinline__ u16x8 lload8(const u16* p) {
    return *(const u16x8*)p;   // 16-byte load (ds_load_b128 when p is LDS)
}

#define BATCH 8
#define TT    2048
#define DM    512
#define DIN   64

// LDS staging geometry (bytes). Rows padded 64B -> 80B for conflict-free ds_load_b128.
#define AROWS   64
#define BROWS   128
#define ROWB    80
#define ABYTES  (AROWS * ROWB)            // 5120
#define BUFBYTES (ABYTES + BROWS * ROWB)  // 5120 + 10240 = 15360
#define BUFELEMS (BUFBYTES / 2)           // 7680 u16

// ---------------- elementwise converts ----------------
__global__ void f2bf_kernel(const float* __restrict__ in, __bf16* __restrict__ out, int n) {
    int i = blockIdx.x * blockDim.x + threadIdx.x;
    if (i < n) out[i] = (__bf16)in[i];
}

__global__ void zerobf_kernel(__bf16* __restrict__ p, int n) {
    int i = blockIdx.x * blockDim.x + threadIdx.x;
    if (i < n) p[i] = (__bf16)0.f;
}

// [cout][cin][ks] f32 -> [ks][cout][cin] bf16
__global__ void reorder_w_kernel(const float* __restrict__ w, __bf16* __restrict__ o,
                                 int cout, int cin, int ks) {
    int i = blockIdx.x * blockDim.x + threadIdx.x;
    int total = cout * cin * ks;
    if (i >= total) return;
    int k  = i % ks;
    int ci = (i / ks) % cin;
    int co = i / (ks * cin);
    o[((size_t)k * cout + co) * cin + ci] = (__bf16)w[i];
}

// ---------------- WMMA causal dilated conv / GEMM, async double-buffered ----
// Block = 128 threads (4 waves). Block tile = 64 t x 128 ch.
// Wave (tw,nw) computes a 32 t x 64 ch sub-tile (2x4 f32 accumulators).
// Stage s = (tap, k0): A[64 x 32ci] + B[128 x 32ci] copied global->LDS with
// global_load_async_to_lds_b128 (6 async ops/wave/stage). Double-buffered:
// issue stage s+1, then s_wait_asynccnt 6 proves stage s landed (in-order).
template<int CIN, int TAPS>
__global__ __launch_bounds__(128)
void conv_wmma_kernel(const __bf16* __restrict__ in,
                      const __bf16* __restrict__ wb,
                      const float*  __restrict__ bias,
                      const __bf16* __restrict__ resid,
                      const __bf16* __restrict__ zrow,   // >=64B of zeros
                      __bf16* __restrict__ out,
                      int dil, int T_, int COUT, int doGelu)
{
    __shared__ __align__(16) u16 smem[2 * BUFELEMS];

    const int tid  = threadIdx.x;
    const int wid  = tid >> 5;
    const int lane = tid & 31;
    const int m    = lane & 15;
    const int hi   = lane >> 4;      // 0: K {0..7,16..23}, 1: K {8..15,24..31}
    const int tw   = wid & 1;        // time half of block tile
    const int nw   = wid >> 1;       // channel half of block tile
    const int t0   = blockIdx.x * 64;
    const int n0   = blockIdx.y * 128;
    const int b    = blockIdx.z;

    const unsigned lds_base = (unsigned)(size_t)((lds_u16*)smem);
    const unsigned long long inb = (unsigned long long)(size_t)in;
    const unsigned long long wbb = (unsigned long long)(size_t)wb;
    const unsigned long long zrb = (unsigned long long)(size_t)zrow;

    constexpr int KSTEPS = CIN / 32;
    constexpr int STAGES = TAPS * KSTEPS;

    auto issueStage = [&](int s, int p) {
        const int tap = s / KSTEPS;
        const int k0  = (s - tap * KSTEPS) * 32;
        const unsigned bufoff = (unsigned)p * BUFBYTES;
        // ---- A: 64 rows, 4 x 16B chunks per row ----
#pragma unroll
        for (int e = 0; e < 2; ++e) {
            const int idx = e * 128 + tid;
            const int row = idx >> 2, c = idx & 3;
            const int trow = t0 + row - tap * dil;
            const unsigned long long g = (trow >= 0)
                ? inb + (((size_t)(b * T_ + trow) * CIN + k0) * 2 + c * 16)
                : zrb + c * 16;
            const unsigned l = lds_base + bufoff + row * ROWB + c * 16;
            asm volatile("global_load_async_to_lds_b128 %0, %1, off"
                         :: "v"(l), "v"(g) : "memory");
        }
        // ---- B: 128 rows, 4 x 16B chunks per row ----
#pragma unroll
        for (int e = 0; e < 4; ++e) {
            const int idx = e * 128 + tid;
            const int row = idx >> 2, c = idx & 3;
            const unsigned long long g =
                wbb + ((((size_t)tap * COUT + n0 + row) * CIN + k0) * 2 + c * 16);
            const unsigned l = lds_base + bufoff + ABYTES + row * ROWB + c * 16;
            asm volatile("global_load_async_to_lds_b128 %0, %1, off"
                         :: "v"(l), "v"(g) : "memory");
        }
    };

    v8f acc[2][4];
#pragma unroll
    for (int i = 0; i < 2; ++i)
#pragma unroll
        for (int j = 0; j < 4; ++j) acc[i][j] = (v8f){};

    issueStage(0, 0);
#pragma unroll 2
    for (int s = 0; s < STAGES; ++s) {
        if (s + 1 < STAGES) {
            issueStage(s + 1, (s + 1) & 1);
            asm volatile("s_wait_asynccnt 6" ::: "memory");  // stage s landed
        } else {
            asm volatile("s_wait_asynccnt 0" ::: "memory");
        }
        __syncthreads();

        const u16* Abase = smem + (s & 1) * BUFELEMS;        // rows of 40 u16
        const u16* Bbase = Abase + ABYTES / 2;
        ABf A[2];
#pragma unroll
        for (int i = 0; i < 2; ++i) {
            const int tr = tw * 32 + i * 16 + m;
            A[i].h[0] = lload8(Abase + tr * 40 + hi * 8);
            A[i].h[1] = lload8(Abase + tr * 40 + 16 + hi * 8);
        }
#pragma unroll
        for (int j = 0; j < 4; ++j) {
            const int brow = nw * 64 + j * 16 + m;
            ABf Bv;
            Bv.h[0] = lload8(Bbase + brow * 40 + hi * 8);
            Bv.h[1] = lload8(Bbase + brow * 40 + 16 + hi * 8);
#pragma unroll
            for (int i = 0; i < 2; ++i)
                acc[i][j] = __builtin_amdgcn_wmma_f32_16x16x32_bf16(
                    false, A[i].v, false, Bv.v, (short)0, acc[i][j], false, false);
        }
        __syncthreads();   // all waves done reading before buffer reuse at s+2
    }

    // ---- epilogue: C/D layout -> lane holds N=m, VGPR r holds M = hi*8+r ----
#pragma unroll
    for (int j = 0; j < 4; ++j) {
        const int   ch = n0 + nw * 64 + j * 16 + m;
        const float bv = bias[ch];
#pragma unroll
        for (int i = 0; i < 2; ++i) {
#pragma unroll
            for (int r = 0; r < 8; ++r) {
                const int t = t0 + tw * 32 + i * 16 + hi * 8 + r;
                float x = acc[i][j][r] + bv;
                if (doGelu) x = 0.5f * x * (1.0f + erff(x * 0.70710678118654752f));
                size_t oidx = ((size_t)(b * T_ + t)) * COUT + ch;
                if (resid) x += (float)resid[oidx];
                out[oidx] = (__bf16)x;
            }
        }
    }
}

// ---------------- attention tail (tiny) ----------------
__global__ void qproj_kernel(const __bf16* __restrict__ z, const float* __restrict__ wq,
                             const float* __restrict__ bq, float* __restrict__ q,
                             int T_, int Dm) {
    int o = blockIdx.x * blockDim.x + threadIdx.x;
    if (o >= BATCH * Dm) return;
    int b = o / Dm, d = o % Dm;
    const __bf16* zr = z + ((size_t)(b * T_ + T_ - 1)) * Dm;
    const float*  wr = wq + (size_t)d * Dm;
    float s = bq[d];
    for (int i = 0; i < Dm; ++i) s += (float)zr[i] * wr[i];
    q[o] = s;
}

__global__ void scores_kernel(const float* __restrict__ q, const __bf16* __restrict__ k,
                              float* __restrict__ sc, int T_, int Dm) {
    int o = blockIdx.x * blockDim.x + threadIdx.x;
    if (o >= BATCH * T_) return;
    int b = o / T_, t = o % T_;
    const float*  qr = q + (size_t)b * Dm;
    const __bf16* kr = k + ((size_t)(b * T_ + t)) * Dm;
    float s = 0.f;
    for (int i = 0; i < Dm; ++i) s += qr[i] * (float)kr[i];
    sc[o] = s * 0.04419417382415922f; // 1/sqrt(512)
}

// top-k(=512) selection with exact top_k tie semantics (stable by index),
// then softmax over kept entries; others get weight 0.
__global__ __launch_bounds__(512)
void topp_softmax_kernel(const float* __restrict__ sc, float* __restrict__ w,
                         int T_, int kkeep) {
    __shared__ float s[TT];
    __shared__ float red[512];
    const int b = blockIdx.x, tid = threadIdx.x;
    const float* sr = sc + (size_t)b * T_;
    for (int i = tid; i < T_; i += 512) s[i] = sr[i];
    __syncthreads();

    float vals[4];
    int   keep[4];
    float lm = -__builtin_inff();
#pragma unroll
    for (int u = 0; u < 4; ++u) {
        const int   i  = tid + u * 512;
        const float si = s[i];
        int rank = 0;
        for (int j = 0; j < T_; ++j) {
            float sj = s[j];
            rank += (sj > si) || (sj == si && j < i);
        }
        keep[u] = (rank < kkeep);
        vals[u] = si;
        if (keep[u] && si > lm) lm = si;
    }
    red[tid] = lm;
    __syncthreads();
    for (int o = 256; o > 0; o >>= 1) {
        if (tid < o) red[tid] = fmaxf(red[tid], red[tid + o]);
        __syncthreads();
    }
    const float mx = red[0];
    __syncthreads();

    float ls = 0.f, ex[4];
#pragma unroll
    for (int u = 0; u < 4; ++u) {
        ex[u] = keep[u] ? __expf(vals[u] - mx) : 0.f;
        ls += ex[u];
    }
    red[tid] = ls;
    __syncthreads();
    for (int o = 256; o > 0; o >>= 1) {
        if (tid < o) red[tid] += red[tid + o];
        __syncthreads();
    }
    const float inv = 1.f / red[0];
#pragma unroll
    for (int u = 0; u < 4; ++u) w[(size_t)b * T_ + tid + u * 512] = ex[u] * inv;
}

__global__ void attnout_kernel(const float* __restrict__ w, const __bf16* __restrict__ v,
                               float* __restrict__ ao, int T_, int Dm) {
    int o = blockIdx.x * blockDim.x + threadIdx.x;
    if (o >= BATCH * Dm) return;
    int b = o / Dm, d = o % Dm;
    const float*  wr = w + (size_t)b * T_;
    const __bf16* vb = v + (size_t)b * T_ * Dm + d;
    float s = 0.f;
    for (int t = 0; t < T_; ++t) s += wr[t] * (float)vb[(size_t)t * Dm];
    ao[o] = s;
}

__global__ void final_kernel(const __bf16* __restrict__ z, const float* __restrict__ ao,
                             const float* __restrict__ wp, const float* __restrict__ bp,
                             float* __restrict__ out, int T_, int Dm) {
    int o = blockIdx.x * blockDim.x + threadIdx.x;
    if (o >= BATCH * Dm) return;
    int b = o / Dm, d = o % Dm;
    const __bf16* zr = z + ((size_t)(b * T_ + T_ - 1)) * Dm;
    const float*  ar = ao + (size_t)b * Dm;
    const float*  wr = wp + (size_t)d * Dm;
    float s = bp[d];
    for (int i = 0; i < Dm; ++i) s += ((float)zr[i] + ar[i]) * wr[i];
    out[o] = s;
}

// ---------------- host orchestration ----------------
extern "C" void kernel_launch(void* const* d_in, const int* in_sizes, int n_in,
                              void* d_out, int out_size, void* d_ws, size_t ws_size,
                              hipStream_t stream) {
    (void)in_sizes; (void)n_in; (void)out_size; (void)ws_size;

    const float* x    = (const float*)d_in[0];
    const float* c1w0 = (const float*)d_in[1];
    const float* c1b0 = (const float*)d_in[2];
    const float* c2w0 = (const float*)d_in[3];
    const float* c2b0 = (const float*)d_in[4];
    const float* resw = (const float*)d_in[5];
    const float* resb = (const float*)d_in[6];
    const float* c1w  = (const float*)d_in[7];
    const float* c1b  = (const float*)d_in[8];
    const float* c2w  = (const float*)d_in[9];
    const float* c2b  = (const float*)d_in[10];
    const float* wq   = (const float*)d_in[11];
    const float* bq   = (const float*)d_in[12];
    const float* wk   = (const float*)d_in[13];
    const float* bk   = (const float*)d_in[14];
    const float* wv   = (const float*)d_in[15];
    const float* bv   = (const float*)d_in[16];
    const float* wp   = (const float*)d_in[17];
    const float* bp   = (const float*)d_in[18];

    size_t off = 0;
    char* base = (char*)d_ws;
    auto alloc = [&](size_t bytes) {
        void* p = base + off;
        off += (bytes + 255) & ~(size_t)255;
        return p;
    };

    __bf16* zrow  = (__bf16*)alloc(256);
    __bf16* xbf   = (__bf16*)alloc((size_t)2 * BATCH * TT * DIN);
    __bf16* hA    = (__bf16*)alloc((size_t)2 * BATCH * TT * DM);
    __bf16* hB    = (__bf16*)alloc((size_t)2 * BATCH * TT * DM);
    __bf16* hC    = (__bf16*)alloc((size_t)2 * BATCH * TT * DM);
    __bf16* c1w0r = (__bf16*)alloc((size_t)2 * 3 * DM * DIN);
    __bf16* c2w0r = (__bf16*)alloc((size_t)2 * 3 * DM * DM);
    __bf16* reswb = (__bf16*)alloc((size_t)2 * DM * DIN);
    __bf16* c1wr  = (__bf16*)alloc((size_t)2 * 5 * 3 * DM * DM);
    __bf16* c2wr  = (__bf16*)alloc((size_t)2 * 5 * 3 * DM * DM);
    __bf16* wkb   = (__bf16*)alloc((size_t)2 * DM * DM);
    __bf16* wvb   = (__bf16*)alloc((size_t)2 * DM * DM);
    float*  qbuf  = (float*)alloc((size_t)4 * BATCH * DM);
    float*  scbuf = (float*)alloc((size_t)4 * BATCH * TT);
    float*  awbuf = (float*)alloc((size_t)4 * BATCH * TT);
    float*  aobuf = (float*)alloc((size_t)4 * BATCH * DM);

    auto cdiv = [](int a, int b) { return (a + b - 1) / b; };

    // ---- weight / input conversion ----
    zerobf_kernel<<<1, 128, 0, stream>>>(zrow, 128);
    {
        int n = BATCH * TT * DIN;
        f2bf_kernel<<<cdiv(n, 256), 256, 0, stream>>>(x, xbf, n);
    }
    reorder_w_kernel<<<cdiv(DM * DIN * 3, 256), 256, 0, stream>>>(c1w0, c1w0r, DM, DIN, 3);
    reorder_w_kernel<<<cdiv(DM * DM * 3, 256), 256, 0, stream>>>(c2w0, c2w0r, DM, DM, 3);
    f2bf_kernel<<<cdiv(DM * DIN, 256), 256, 0, stream>>>(resw, reswb, DM * DIN);
    for (int i = 0; i < 5; ++i) {
        reorder_w_kernel<<<cdiv(DM * DM * 3, 256), 256, 0, stream>>>(
            c1w + (size_t)i * DM * DM * 3, c1wr + (size_t)i * 3 * DM * DM, DM, DM, 3);
        reorder_w_kernel<<<cdiv(DM * DM * 3, 256), 256, 0, stream>>>(
            c2w + (size_t)i * DM * DM * 3, c2wr + (size_t)i * 3 * DM * DM, DM, DM, 3);
    }
    f2bf_kernel<<<cdiv(DM * DM, 256), 256, 0, stream>>>(wk, wkb, DM * DM);
    f2bf_kernel<<<cdiv(DM * DM, 256), 256, 0, stream>>>(wv, wvb, DM * DM);

    // ---- conv stack ----
    dim3 cgrid(TT / 64, DM / 128, BATCH);
    // block 0
    conv_wmma_kernel<DIN, 3><<<cgrid, 128, 0, stream>>>(xbf, c1w0r, c1b0, nullptr, zrow, hB, 1, TT, DM, 1);
    conv_wmma_kernel<DM, 3><<<cgrid, 128, 0, stream>>>(hB, c2w0r, c2b0, nullptr, zrow, hC, 1, TT, DM, 1);
    // h = y2 + (x @ resw + resb)   (1x1 conv, residual add fused)
    conv_wmma_kernel<DIN, 1><<<cgrid, 128, 0, stream>>>(xbf, reswb, resb, hC, zrow, hA, 1, TT, DM, 0);
    // blocks 1..5
    __bf16* cur = hA;
    __bf16* alt = hC;
    for (int i = 0; i < 5; ++i) {
        int dil = 2 << i;
        conv_wmma_kernel<DM, 3><<<cgrid, 128, 0, stream>>>(
            cur, c1wr + (size_t)i * 3 * DM * DM, c1b + (size_t)i * DM, nullptr, zrow, hB, dil, TT, DM, 1);
        conv_wmma_kernel<DM, 3><<<cgrid, 128, 0, stream>>>(
            hB, c2wr + (size_t)i * 3 * DM * DM, c2b + (size_t)i * DM, cur, zrow, alt, dil, TT, DM, 1);
        __bf16* t = cur; cur = alt; alt = t;
    }
    __bf16* z = cur; // final hidden state [B,T,D] bf16

    // ---- k/v projections (WMMA GEMM, taps=1) ----
    conv_wmma_kernel<DM, 1><<<cgrid, 128, 0, stream>>>(z, wkb, bk, nullptr, zrow, hB, 1, TT, DM, 0);
    conv_wmma_kernel<DM, 1><<<cgrid, 128, 0, stream>>>(z, wvb, bv, nullptr, zrow, alt, 1, TT, DM, 0);
    __bf16* kbuf = hB;
    __bf16* vbuf = alt;

    // ---- attention tail ----
    qproj_kernel<<<cdiv(BATCH * DM, 256), 256, 0, stream>>>(z, wq, bq, qbuf, TT, DM);
    scores_kernel<<<cdiv(BATCH * TT, 256), 256, 0, stream>>>(qbuf, kbuf, scbuf, TT, DM);
    topp_softmax_kernel<<<BATCH, 512, 0, stream>>>(scbuf, awbuf, TT, TT / 4);
    attnout_kernel<<<cdiv(BATCH * DM, 256), 256, 0, stream>>>(awbuf, vbuf, aobuf, TT, DM);
    final_kernel<<<cdiv(BATCH * DM, 256), 256, 0, stream>>>(z, aobuf, wp, bp, (float*)d_out, TT, DM);
}
